// DenseEquiMessageBlock_34376918237211
// MI455X (gfx1250) — matrix operand order repaired and checked by reading, not compiled, not run on hardware
//
#include <hip/hip_runtime.h>
#include <hip/hip_bf16.h>

typedef __attribute__((ext_vector_type(16))) _Float16 v16h;
typedef __attribute__((ext_vector_type(8)))  _Float16 v8h;
typedef __attribute__((ext_vector_type(8)))  float    v8f;

constexpr int FD = 128;   // feature dim
constexpr int ND = 384;   // nodes
constexpr int BD = 2;     // batch
constexpr int TJ = 16;    // receiver tile per block
constexpr int TI = 8;     // sender chunk per iteration
constexpr int NE = TI * TJ;    // 128 edges per iteration
constexpr int SPLIT = 4;       // sender-range splits (4x grid parallelism)
constexpr int ISPAN = ND / SPLIT;  // 96 senders per split

// ---------------- WMMA fragment helpers (contiguous k = 16*hi + t, same order for A and B) ----------

// A: 16x32 f16. lane m = lane&15; K = 32*kc + 16*hi + t (contiguous)
__device__ __forceinline__ v16h load_a_frag(const _Float16* X, int mBase, int lane, int kc) {
  int m  = mBase + (lane & 15);
  int hi = lane >> 4;
  const _Float16* p = X + m * FD + kc * 32 + hi * 16;
  return *(const v16h*)p;
}

// B: 32x16 f16 from transposed weight Wt[out][k]. col n = lane&15,
// K = 32*kc + 16*hi + t  -> one contiguous 32B load per lane.
__device__ __forceinline__ void load_bfrags(const _Float16* Wt, int nBase, int lane,
                                            v16h bf[4]) {
  int n  = nBase + (lane & 15);
  int hi = lane >> 4;
#pragma unroll
  for (int kc = 0; kc < 4; ++kc)
    bf[kc] = *(const v16h*)(Wt + n * FD + kc * 32 + hi * 16);
}

// 16x16 tile of X[M x 128] @ W (B-fragments preloaded in registers)
__device__ __forceinline__ v8f wmma_tile_pre(const _Float16* X, int mBase,
                                             const v16h bf[4], int lane) {
  v8f acc = {};
#pragma unroll
  for (int kc = 0; kc < 4; ++kc) {
    v16h a = load_a_frag(X, mBase, lane, kc);
    acc = __builtin_amdgcn_wmma_f32_16x16x32_f16(false, a, false, bf[kc], (short)0, acc,
                                                 false, false);
  }
  return acc;
}

// single fast-rcp silu: x * v_rcp_f32(1+exp(-x))
__device__ __forceinline__ float silu_f(float x) {
  return x * __builtin_amdgcn_rcpf(1.f + __expf(-x));
}

// ---------------- Kernel 1: weight transpose+convert f32[k][out] -> f16[out][k] --
__global__ void eqmb_wprep_kernel(const float* __restrict__ W, _Float16* __restrict__ Wt,
                                  int OUT, int total) {
  int idx = blockIdx.x * blockDim.x + threadIdx.x;
  if (idx < total) {
    int n = idx >> 7;       // out index
    int k = idx & 127;      // k index
    Wt[idx] = (_Float16)W[k * OUT + n];
  }
}

// ---------------- Kernel 2: phi = silu(h@W1+b1)@W2+b2 ---------------------------
__global__ void __launch_bounds__(256)
eqmb_phi_kernel(const float* __restrict__ h,
                const _Float16* __restrict__ W1t, const float* __restrict__ b1,
                const _Float16* __restrict__ W2t, const float* __restrict__ b2,
                float* __restrict__ phi) {
  __shared__ _Float16 Xh[TJ * FD];
  __shared__ _Float16 Yh[TJ * FD];
  const int tid = threadIdx.x, wave = tid >> 5, lane = tid & 31;
  const int row0 = blockIdx.x * TJ;  // 48 blocks cover 768 rows

  for (int idx = tid; idx < TJ * FD; idx += 256)
    Xh[idx] = (_Float16)h[row0 * FD + idx];
  __syncthreads();

  {  // layer 1
    int nBase = wave * 16;
    int n = nBase + (lane & 15), hi = lane >> 4;
    float bb = b1[n];
    v16h bf[4];
    load_bfrags(W1t, nBase, lane, bf);
    v8f acc = wmma_tile_pre(Xh, 0, bf, lane);
#pragma unroll
    for (int r = 0; r < 8; ++r) {
      int m = r + hi * 8;
      Yh[m * FD + n] = (_Float16)silu_f(acc[r] + bb);
    }
  }
  __syncthreads();
  {  // layer 2 -> global f32
    int nBase = wave * 16;
    int n = nBase + (lane & 15), hi = lane >> 4;
    float bb = b2[n];
    v16h bf[4];
    load_bfrags(W2t, nBase, lane, bf);
    v8f acc = wmma_tile_pre(Yh, 0, bf, lane);
#pragma unroll
    for (int r = 0; r < 8; ++r) {
      int m = r + hi * 8;
      phi[(row0 + m) * FD + n] = acc[r] + bb;
    }
  }
}

// ---------------- Kernel 3: fused per-edge MLP chain + reductions ---------------
// dynamic LDS layout (bytes)
constexpr int OFF_XA   = 0;        // 128x128 f16 = 32768
constexpr int OFF_XB   = 32768;    // 32768
constexpr int OFF_ADV  = 65536;    // accDV 16*128*3 f32 = 24576
constexpr int OFF_AFH  = 90112;    // accFH 16*128 f32 = 8192
constexpr int OFF_PHJ  = 98304;    // phiJ 16*128 f32 = 8192
constexpr int OFF_PHI  = 106496;   // phiI 8*128 f32 = 4096
constexpr int OFF_VI   = 110592;   // vI 8*128*3 f32 = 12288
constexpr int OFF_UNIT = 122880;   // unit 128*3 f32 = 1536
constexpr int OFF_ADJ  = 124416;   // 128 f32 = 512
constexpr int OFF_D    = 124928;   // 128 f32 = 512
constexpr int OFF_OFFS = 125440;   // 128 f32 = 512
constexpr int OFF_XYZ  = 125952;   // xyzJ 48 + xyzI 24 f32
constexpr int SMEM_BYTES = 126976;

__global__ void __launch_bounds__(256, 1)
eqmb_edge_kernel(const float* __restrict__ v, const float* __restrict__ adj,
                 const float* __restrict__ xyz,
                 const _Float16* __restrict__ W3t, const _Float16* __restrict__ W4t,
                 const _Float16* __restrict__ W5t, const _Float16* __restrict__ W6t,
                 const float* __restrict__ b3, const float* __restrict__ b4,
                 const float* __restrict__ b5, const float* __restrict__ b6,
                 const float* __restrict__ offset, const float* __restrict__ phi,
                 float* __restrict__ accFHg, float* __restrict__ dvPart) {
  extern __shared__ char smem[];
  _Float16* Xa  = (_Float16*)(smem + OFF_XA);
  _Float16* Xb  = (_Float16*)(smem + OFF_XB);
  float* accDV  = (float*)(smem + OFF_ADV);
  float* accFH  = (float*)(smem + OFF_AFH);
  float* phiJs  = (float*)(smem + OFF_PHJ);
  float* phiIs  = (float*)(smem + OFF_PHI);
  float* vIs    = (float*)(smem + OFF_VI);
  float* unitS  = (float*)(smem + OFF_UNIT);
  float* adjS   = (float*)(smem + OFF_ADJ);
  float* dS     = (float*)(smem + OFF_D);
  float* offS   = (float*)(smem + OFF_OFFS);
  float* xyzJ   = (float*)(smem + OFF_XYZ);
  float* xyzI   = xyzJ + TJ * 3;

  const int tid = threadIdx.x;
  const int wave = tid >> 5, lane = tid & 31;
  const int b  = blockIdx.x / (ND / TJ);
  const int j0 = (blockIdx.x % (ND / TJ)) * TJ;
  const int split = blockIdx.y;            // sender-range split for grid parallelism
  const int iBeg = split * ISPAN;
  const int iEnd = iBeg + ISPAN;

  for (int idx = tid; idx < TJ * FD * 3; idx += 256) accDV[idx] = 0.f;
  for (int idx = tid; idx < TJ * FD; idx += 256) accFH[idx] = 0.f;
  if (tid < FD) offS[tid] = offset[tid];
  if (tid < TJ * 3) xyzJ[tid] = xyz[(b * ND + j0) * 3 + tid];
  for (int idx = tid; idx < TJ * FD; idx += 256) {
    int jj = idx >> 7, f = idx & 127;
    phiJs[idx] = phi[(b * ND + j0 + jj) * FD + f];
  }
  __syncthreads();

  for (int i0 = iBeg; i0 < iEnd; i0 += TI) {
    // prefetch next chunk (global_prefetch_b8) while this one computes
    if (i0 + TI < iEnd) {
      const float* pv = v + (b * ND + i0 + TI) * FD * 3;
      const float* pp = phi + (b * ND + i0 + TI) * FD;
      if (tid < TI * 3) __builtin_prefetch(pv + tid * FD, 0, 1);
      if (tid < TI)     __builtin_prefetch(pp + tid * FD, 0, 1);
    }
    // A1: load sender-chunk data
    if (tid < TI * 3) xyzI[tid] = xyz[(b * ND + i0) * 3 + tid];
    for (int idx = tid; idx < TI * FD * 3; idx += 256)
      vIs[idx] = v[(b * ND + i0) * FD * 3 + idx];
    for (int idx = tid; idx < TI * FD; idx += 256)
      phiIs[idx] = phi[(b * ND + i0) * FD + idx];
    __syncthreads();
    // A2: geometry for the 8x16 pair block
    if (tid < NE) {
      int ii = tid >> 4, jj = tid & 15;
      int i = i0 + ii, j = j0 + jj;
      float rx = xyzJ[jj * 3 + 0] - xyzI[ii * 3 + 0];
      float ry = xyzJ[jj * 3 + 1] - xyzI[ii * 3 + 1];
      float rz = xyzJ[jj * 3 + 2] - xyzI[ii * 3 + 2];
      float d2 = rx * rx + ry * ry + rz * rz;
      float d  = d2 > 0.f ? sqrtf(d2) : 0.f;
      dS[tid] = d;
      float den = d + (i == j ? 1.f : 0.f);
      float inv = (den != 0.f) ? __builtin_amdgcn_rcpf(den) : 0.f;
      unitS[tid * 3 + 0] = rx * inv;
      unitS[tid * 3 + 1] = ry * inv;
      unitS[tid * 3 + 2] = rz * inv;
      adjS[tid] = adj[(b * ND + i) * ND + j];
    }
    __syncthreads();
    // A3: rbf -> Xa (f16), generated on the fly (never touches HBM)
    for (int idx = tid; idx < NE * FD; idx += 256) {
      int m = idx >> 7, k = idx & 127;
      float t = dS[m] - offS[k];
      Xa[idx] = (_Float16)__expf(-t * t);
    }
    __syncthreads();
    // L1: Xb = silu(Xa @ W3 + b3)
    {
      int nBase = wave * 16;
      int n = nBase + (lane & 15), hi = lane >> 4;
      float bb = b3[n];
      v16h bf[4];
      load_bfrags(W3t, nBase, lane, bf);
#pragma unroll
      for (int mt = 0; mt < 8; ++mt) {
        v8f acc = wmma_tile_pre(Xa, mt * 16, bf, lane);
#pragma unroll
        for (int r = 0; r < 8; ++r) {
          int m = mt * 16 + r + hi * 8;
          Xb[m * FD + n] = (_Float16)silu_f(acc[r] + bb);
        }
      }
    }
    __syncthreads();
    // L2: Xa = (Xb @ W4 + b4) * phi_i * phi_j
    {
      int nBase = wave * 16;
      int n = nBase + (lane & 15), hi = lane >> 4;
      float bb = b4[n];
      v16h bf[4];
      load_bfrags(W4t, nBase, lane, bf);
#pragma unroll
      for (int mt = 0; mt < 8; ++mt) {
        v8f acc = wmma_tile_pre(Xb, mt * 16, bf, lane);
#pragma unroll
        for (int r = 0; r < 8; ++r) {
          int m = mt * 16 + r + hi * 8;
          int ii = m >> 4, jj = m & 15;
          float x = (acc[r] + bb) * phiIs[ii * FD + n] * phiJs[jj * FD + n];
          Xa[m * FD + n] = (_Float16)x;
        }
      }
    }
    __syncthreads();
    // L3: Xb = silu(Xa @ W5 + b5)
    {
      int nBase = wave * 16;
      int n = nBase + (lane & 15), hi = lane >> 4;
      float bb = b5[n];
      v16h bf[4];
      load_bfrags(W5t, nBase, lane, bf);
#pragma unroll
      for (int mt = 0; mt < 8; ++mt) {
        v8f acc = wmma_tile_pre(Xa, mt * 16, bf, lane);
#pragma unroll
        for (int r = 0; r < 8; ++r) {
          int m = mt * 16 + r + hi * 8;
          Xb[m * FD + n] = (_Float16)silu_f(acc[r] + bb);
        }
      }
    }
    __syncthreads();
    // L4: fw = Xb @ W6 + b6 (384 outs); mask by adj; reduce over ii in REGISTERS.
    // m = mt*16 + r + hi*8  =>  ii == mt, jj == r + hi*8: the target (jj,f) is
    // mt-invariant, so the 8-way ii sum lives in registers; one atomic set per r.
    {
#pragma unroll
      for (int nn = 0; nn < 3; ++nn) {
        int nBase = (wave + nn * 8) * 16;  // waves cover all 24 out-tiles
        int o = nBase + (lane & 15), hi = lane >> 4;
        int c = o % 3, f = o / 3;          // fw.reshape(...,F,3): out = 3f+c
        float bb = b6[o];
        v16h bf[4];
        load_bfrags(W6t, nBase, lane, bf);
        float s0[8], s1[8], s2[8];
#pragma unroll
        for (int r = 0; r < 8; ++r) { s0[r] = 0.f; s1[r] = 0.f; s2[r] = 0.f; }
#pragma unroll
        for (int mt = 0; mt < 8; ++mt) {
          v8f acc = wmma_tile_pre(Xb, mt * 16, bf, lane);
          const int ii = mt;
#pragma unroll
          for (int r = 0; r < 8; ++r) {
            int jj = r + hi * 8;
            float val = (acc[r] + bb) * adjS[ii * 16 + jj];
            float m0 = (c == 0) ? unitS[(ii * 16 + jj) * 3 + 0]
                     : (c == 1) ? vIs[(ii * FD + f) * 3 + 0] : 1.f;
            float m1 = (c == 0) ? unitS[(ii * 16 + jj) * 3 + 1]
                     : (c == 1) ? vIs[(ii * FD + f) * 3 + 1] : 0.f;
            float m2 = (c == 0) ? unitS[(ii * 16 + jj) * 3 + 2]
                     : (c == 1) ? vIs[(ii * FD + f) * 3 + 2] : 0.f;
            s0[r] += val * m0;
            s1[r] += val * m1;
            s2[r] += val * m2;
          }
        }
#pragma unroll
        for (int r = 0; r < 8; ++r) {
          int jj = r + hi * 8;
          if (c == 2) {
            atomicAdd(&accFH[jj * FD + f], s0[r]);
          } else {
            atomicAdd(&accDV[(jj * FD + f) * 3 + 0], s0[r]);
            atomicAdd(&accDV[(jj * FD + f) * 3 + 1], s1[r]);
            atomicAdd(&accDV[(jj * FD + f) * 3 + 2], s2[r]);
          }
        }
      }
    }
    __syncthreads();
  }

  // write per-split partials (deterministic, no global atomics)
  float* fhDst = accFHg + split * (BD * ND * FD);
  float* dvDst = dvPart + split * (BD * ND * FD * 3);
  for (int idx = tid; idx < TJ * FD; idx += 256) {
    int jj = idx >> 7, f = idx & 127;
    fhDst[(b * ND + j0 + jj) * FD + f] = accFH[idx];
  }
  for (int idx = tid; idx < TJ * FD * 3; idx += 256) {
    int jj = idx / (FD * 3), rem = idx % (FD * 3);
    dvDst[(b * ND + j0 + jj) * FD * 3 + rem] = accDV[idx];
  }
}

// ---------------- Kernel 4a: dv = sum of split partials -------------------------
__global__ void eqmb_dvred_kernel(const float* __restrict__ dvPart,
                                  float* __restrict__ dv) {
  int idx = blockIdx.x * blockDim.x + threadIdx.x;
  if (idx < BD * ND * FD * 3) {
    float s = 0.f;
#pragma unroll
    for (int p = 0; p < SPLIT; ++p) s += dvPart[p * (BD * ND * FD * 3) + idx];
    dv[idx] = s;
  }
}

// ---------------- Kernel 4b: dh = LN((sum of fh partials) * h) ------------------
__global__ void __launch_bounds__(128)
eqmb_ln_kernel(const float* __restrict__ accFHg, const float* __restrict__ h,
               const float* __restrict__ gamma, const float* __restrict__ beta,
               float* __restrict__ dh) {
  __shared__ float red[FD];
  int row = blockIdx.x, f = threadIdx.x;
  float s = 0.f;
#pragma unroll
  for (int p = 0; p < SPLIT; ++p) s += accFHg[p * (BD * ND * FD) + row * FD + f];
  float val = s * h[row * FD + f];
  red[f] = val;
  __syncthreads();
  for (int st = 64; st > 0; st >>= 1) { if (f < st) red[f] += red[f + st]; __syncthreads(); }
  float mu = red[0] * (1.f / FD);
  __syncthreads();
  float dvv = val - mu;
  red[f] = dvv * dvv;
  __syncthreads();
  for (int st = 64; st > 0; st >>= 1) { if (f < st) red[f] += red[f + st]; __syncthreads(); }
  float var = red[0] * (1.f / FD);
  dh[row * FD + f] = dvv * __builtin_amdgcn_rsqf(var + 1e-5f) * gamma[f] + beta[f];
}

// ---------------- launch ---------------------------------------------------------
extern "C" void kernel_launch(void* const* d_in, const int* in_sizes, int n_in,
                              void* d_out, int out_size, void* d_ws, size_t ws_size,
                              hipStream_t stream) {
  const float* h    = (const float*)d_in[0];
  const float* v    = (const float*)d_in[1];
  const float* adj  = (const float*)d_in[2];
  const float* xyz  = (const float*)d_in[3];
  const float* W1   = (const float*)d_in[4];
  const float* b1   = (const float*)d_in[5];
  const float* W2   = (const float*)d_in[6];
  const float* b2   = (const float*)d_in[7];
  const float* W3   = (const float*)d_in[8];
  const float* b3   = (const float*)d_in[9];
  const float* W4   = (const float*)d_in[10];
  const float* b4   = (const float*)d_in[11];
  const float* W5   = (const float*)d_in[12];
  const float* b5   = (const float*)d_in[13];
  const float* W6   = (const float*)d_in[14];
  const float* b6   = (const float*)d_in[15];
  const float* gam  = (const float*)d_in[16];
  const float* bet  = (const float*)d_in[17];
  const float* offs = (const float*)d_in[18];

  // workspace layout:
  //   [0,   256KB) f16 transposed weights
  //   [256KB, +384KB) phi (f32)
  //   then SPLIT x accFH partials (4 x 384KB) and SPLIT x dv partials (4 x 1.125MB)
  _Float16* Wt  = (_Float16*)d_ws;
  _Float16* W1t = Wt;
  _Float16* W2t = Wt + 16384;
  _Float16* W3t = Wt + 32768;
  _Float16* W4t = Wt + 49152;
  _Float16* W5t = Wt + 65536;
  _Float16* W6t = Wt + 81920;            // 384*128 halves
  float* phi    = (float*)((char*)d_ws + 262144);
  float* accFHg = phi + BD * ND * FD;
  float* dvPart = accFHg + SPLIT * (BD * ND * FD);

  float* dh = (float*)d_out;
  float* dv = dh + BD * ND * FD;

  // 1) weight prep
  eqmb_wprep_kernel<<<(16384 + 255) / 256, 256, 0, stream>>>(W1, W1t, 128, 16384);
  eqmb_wprep_kernel<<<(16384 + 255) / 256, 256, 0, stream>>>(W2, W2t, 128, 16384);
  eqmb_wprep_kernel<<<(16384 + 255) / 256, 256, 0, stream>>>(W3, W3t, 128, 16384);
  eqmb_wprep_kernel<<<(16384 + 255) / 256, 256, 0, stream>>>(W4, W4t, 128, 16384);
  eqmb_wprep_kernel<<<(16384 + 255) / 256, 256, 0, stream>>>(W5, W5t, 128, 16384);
  eqmb_wprep_kernel<<<(49152 + 255) / 256, 256, 0, stream>>>(W6, W6t, 384, 49152);

  // 2) phi
  eqmb_phi_kernel<<<BD * ND / TJ, 256, 0, stream>>>(h, W1t, b1, W2t, b2, phi);

  // 3) fused edge kernel: 48 j-tiles x 4 sender-splits = 192 blocks,
  //    124KB dynamic LDS each (of the 320KB WGP budget)
  (void)hipFuncSetAttribute((const void*)eqmb_edge_kernel,
                            hipFuncAttributeMaxDynamicSharedMemorySize, SMEM_BYTES);
  dim3 egrid(BD * (ND / TJ), SPLIT);
  eqmb_edge_kernel<<<egrid, 256, SMEM_BYTES, stream>>>(
      v, adj, xyz, W3t, W4t, W5t, W6t, b3, b4, b5, b6, offs, phi, accFHg, dvPart);

  // 4) reductions of split partials
  eqmb_dvred_kernel<<<(BD * ND * FD * 3 + 255) / 256, 256, 0, stream>>>(dvPart, dv);
  eqmb_ln_kernel<<<BD * ND, FD, 0, stream>>>(accFHg, h, gam, bet, dh);
}
